// NonLocalBlock_28235115004046
// MI455X (gfx1250) — matrix operand order repaired and checked by reading, not compiled, not run on hardware
//
#include <hip/hip_runtime.h>

// ---------------------------------------------------------------------------
// NonLocal (SAGAN) block for MI455X / gfx1250.
// B=16, H=W=64, C=128, C_ATTN=16, C_G=64.  N=4096 queries, M=1024 pooled keys.
// All GEMMs via v_wmma_f32_16x16x32_f16 (wave32 WMMA), flash-style fused
// softmax so the [4096,1024] attention matrix never touches HBM.
// ---------------------------------------------------------------------------

typedef _Float16 f16;
typedef __attribute__((ext_vector_type(16))) _Float16 v16h;
typedef __attribute__((ext_vector_type(8)))  _Float16 v8h;
typedef __attribute__((ext_vector_type(8)))  float    v8f;
typedef __attribute__((ext_vector_type(4)))  float    v4f;

__device__ __forceinline__ v8f zero8f() {
  v8f z;
#pragma unroll
  for (int i = 0; i < 8; ++i) z[i] = 0.0f;
  return z;
}
__device__ __forceinline__ v16h zero16h() {
  v16h z;
#pragma unroll
  for (int i = 0; i < 16; ++i) z[i] = (f16)0.0f;
  return z;
}

// ---------------------------------------------------------------------------
// Kernel W: pack/transpose all weights into f16 layouts where every WMMA
// B-fragment is two 8-half loads from one row per lane (fixed N, consecutive K).
//   wTqkg[96][128]  : n<16 -> w_theta col, n<32 -> w_phi col, else w_g col
//   wAT  [128][64]  : wAT[n][k] = w_attn[k][n]
// ---------------------------------------------------------------------------
__global__ __launch_bounds__(256) void wconv(
    const float* __restrict__ wt, const float* __restrict__ wp,
    const float* __restrict__ wg, const float* __restrict__ wa,
    f16* __restrict__ wTqkg, f16* __restrict__ wAT) {
  int i = blockIdx.x * blockDim.x + threadIdx.x;
  if (i < 96 * 128) {
    int n = i / 128, k = i % 128;
    float v = (n < 16) ? wt[k * 16 + n]
            : (n < 32) ? wp[k * 16 + (n - 16)]
                       : wg[k * 64 + (n - 32)];
    wTqkg[i] = (f16)v;
  } else if (i < 96 * 128 + 128 * 64) {
    int j = i - 96 * 128;
    int n = j / 64, k = j % 64;
    wAT[j] = (f16)wa[k * 128 + n];   // transpose of w_attn[64][128]
  }
}

// ---------------------------------------------------------------------------
// Kernel A1: fused QKV projection GEMM  [B*4096,128] x [128,96] -> f16.
// One wave per 16-row tile, full K=32 chunks, 6 N-tiles, 24 WMMAs per wave.
// ---------------------------------------------------------------------------
__global__ __launch_bounds__(32) void proj_qkg(
    const float* __restrict__ x, const f16* __restrict__ wT,
    f16* __restrict__ projAll) {
  const int lane = threadIdx.x;
  const int rsel = lane & 15;
  const bool lo  = lane < 16;
  const long Mbase = (long)blockIdx.x * 16;

  v8f acc[6];
#pragma unroll
  for (int nt = 0; nt < 6; ++nt) acc[nt] = zero8f();

  const float* xr = x + (Mbase + rsel) * 128;

#pragma unroll
  for (int kc = 0; kc < 4; ++kc) {
    // A fragment: lane<16 holds K = kc*32 + {0..7, 16..23}; lane>=16 +8.
    const int k0 = kc * 32 + (lo ? 0 : 8);
    const v4f f0 = *(const v4f*)(xr + k0);
    const v4f f1 = *(const v4f*)(xr + k0 + 4);
    const v4f f2 = *(const v4f*)(xr + k0 + 16);
    const v4f f3 = *(const v4f*)(xr + k0 + 20);
    v16h a;
#pragma unroll
    for (int i = 0; i < 4; ++i) {
      a[i]      = (f16)f0[i];
      a[4 + i]  = (f16)f1[i];
      a[8 + i]  = (f16)f2[i];
      a[12 + i] = (f16)f3[i];
    }
    const int kb = kc * 32 + (lo ? 0 : 8);  // B frag: K = kb+{0..7, 16..23}
#pragma unroll
    for (int nt = 0; nt < 6; ++nt) {
      const f16* wrow = wT + (nt * 16 + rsel) * 128 + kb;
      const v8h b0 = *(const v8h*)(wrow);
      const v8h b1 = *(const v8h*)(wrow + 16);
      v16h bf;
#pragma unroll
      for (int i = 0; i < 8; ++i) { bf[i] = b0[i]; bf[8 + i] = b1[i]; }
      acc[nt] = __builtin_amdgcn_wmma_f32_16x16x32_f16(
          false, a, false, bf, (short)0, acc[nt], false, false);
    }
  }
  // C layout: VGPR r -> row (r + (lane<16?0:8)), col = lane&15
#pragma unroll
  for (int nt = 0; nt < 6; ++nt)
#pragma unroll
    for (int r = 0; r < 8; ++r) {
      const long m = Mbase + r + (lo ? 0 : 8);
      projAll[m * 96 + nt * 16 + rsel] = (f16)acc[nt][r];
    }
}

// ---------------------------------------------------------------------------
// Kernel A2: 2x2 maxpool of phi (cols 16..31) and g (cols 32..95).
//   phiP[b][1024][16]   (row per key -> contiguous B-frag for scores)
//   gT  [b][64][1024]   (transposed  -> contiguous B-frag for P@G)
// ---------------------------------------------------------------------------
__global__ __launch_bounds__(256) void pool_kg(
    const f16* __restrict__ projAll, f16* __restrict__ phiP,
    f16* __restrict__ gT) {
  int i = blockIdx.x * blockDim.x + threadIdx.x;
  if (i >= 16 * 1024 * 80) return;
  const int b  = i / (1024 * 80);
  const int r  = i % (1024 * 80);
  const int pm = r / 80;
  const int d  = r % 80;
  const int ph = pm >> 5, pw = pm & 31;
  const long base = (long)b * 4096 + (ph * 2) * 64 + pw * 2;
  const int c = 16 + d;  // phi cols 16..31 (d<16), g cols 32..95 (d>=16)
  float m0 = (float)projAll[(base)      * 96 + c];
  float m1 = (float)projAll[(base + 1)  * 96 + c];
  float m2 = (float)projAll[(base + 64) * 96 + c];
  float m3 = (float)projAll[(base + 65) * 96 + c];
  float mx = fmaxf(fmaxf(m0, m1), fmaxf(m2, m3));
  if (d < 16) phiP[((long)b * 1024 + pm) * 16 + d] = (f16)mx;
  else        gT[((long)b * 64 + (d - 16)) * 1024 + pm] = (f16)mx;
}

// ---------------------------------------------------------------------------
// Kernel B: flash attention + output projection + residual, fused.
// One wave per 16-query tile; 32 keys/iteration so P@G runs full K=32.
// ---------------------------------------------------------------------------
__global__ __launch_bounds__(32) void attn_fuse(
    const float* __restrict__ x, const f16* __restrict__ projAll,
    const f16* __restrict__ phiP, const f16* __restrict__ gT,
    const f16* __restrict__ wAT, const float* __restrict__ sigmaPtr,
    float* __restrict__ out) {
  __shared__ f16 Pbuf[16 * 32];  // probs tile (C-layout -> A-layout bridge)
  __shared__ f16 Obuf[16 * 64];  // attn_g tile for projection A fragments

  const int lane = threadIdx.x;
  const int rsel = lane & 15;
  const bool lo  = lane < 16;
  const int t    = blockIdx.x;      // B * 256 tiles
  const int b    = t >> 8;
  const long Mb  = (long)(t & 255) * 16;

  // theta A-fragment: K=16 valid, upper half zero-padded.
  v16h aq = zero16h();
  {
    if (lo) {
      const v8h q0 = *(const v8h*)(projAll + ((long)b * 4096 + Mb + rsel) * 96);
      const v8h q1 = *(const v8h*)(projAll + ((long)b * 4096 + Mb + rsel) * 96 + 8);
#pragma unroll
      for (int i = 0; i < 8; ++i) { aq[i] = q0[i]; aq[8 + i] = q1[i]; }
    }
  }

  float mrow[8], lrow[8];
  v8f O[4];
#pragma unroll
  for (int r = 0; r < 8; ++r) { mrow[r] = -3.0e38f; lrow[r] = 0.0f; }
#pragma unroll
  for (int cc = 0; cc < 4; ++cc) O[cc] = zero8f();

  const f16* phiB = phiP + (long)b * 1024 * 16;
  const f16* gB   = gT + (long)b * 64 * 1024;

  for (int kt = 0; kt < 32; ++kt) {
    // ---- scores: S = theta @ phi^T for 32 keys (two 16-key WMMAs)
    v16h bk0 = zero16h(), bk1 = zero16h();
    if (lo) {
      const f16* prow = phiB + (kt * 32 + rsel) * 16;
      const v8h p0 = *(const v8h*)(prow);
      const v8h p1 = *(const v8h*)(prow + 8);
      const v8h p2 = *(const v8h*)(prow + 256);       // key +16
      const v8h p3 = *(const v8h*)(prow + 256 + 8);
#pragma unroll
      for (int i = 0; i < 8; ++i) {
        bk0[i] = p0[i]; bk0[8 + i] = p1[i];
        bk1[i] = p2[i]; bk1[8 + i] = p3[i];
      }
      if (kt + 1 < 32)
        __builtin_prefetch(phiB + ((kt + 1) * 32 + rsel) * 16, 0, 0);
    }
    v8f S0 = __builtin_amdgcn_wmma_f32_16x16x32_f16(
        false, aq, false, bk0, (short)0, zero8f(), false, false);
    v8f S1 = __builtin_amdgcn_wmma_f32_16x16x32_f16(
        false, aq, false, bk1, (short)0, zero8f(), false, false);

    // ---- online softmax per row (16 cols per lane-group, 32 keys total)
#pragma unroll
    for (int r = 0; r < 8; ++r) {
      float s0 = S0[r], s1 = S1[r];
      float g = fmaxf(s0, s1);
      g = fmaxf(g, __shfl_xor(g, 1, 32));
      g = fmaxf(g, __shfl_xor(g, 2, 32));
      g = fmaxf(g, __shfl_xor(g, 4, 32));
      g = fmaxf(g, __shfl_xor(g, 8, 32));
      const float mnew  = fmaxf(mrow[r], g);
      const float scale = __expf(mrow[r] - mnew);
      const float p0 = __expf(s0 - mnew);
      const float p1 = __expf(s1 - mnew);
      float ps = p0 + p1;
      ps += __shfl_xor(ps, 1, 32);
      ps += __shfl_xor(ps, 2, 32);
      ps += __shfl_xor(ps, 4, 32);
      ps += __shfl_xor(ps, 8, 32);
      lrow[r] = lrow[r] * scale + ps;
      mrow[r] = mnew;
#pragma unroll
      for (int cc = 0; cc < 4; ++cc) O[cc][r] *= scale;
      const int M = r + (lo ? 0 : 8);
      Pbuf[M * 32 + rsel]      = (f16)p0;
      Pbuf[M * 32 + 16 + rsel] = (f16)p1;
    }
    __syncthreads();

    // ---- P A-fragment (full K=32) via LDS layout conversion
    v16h ap;
    {
      const v8h a0 = *(const v8h*)(&Pbuf[rsel * 32 + (lo ? 0 : 8)]);
      const v8h a1 = *(const v8h*)(&Pbuf[rsel * 32 + 16 + (lo ? 0 : 8)]);
#pragma unroll
      for (int i = 0; i < 8; ++i) { ap[i] = a0[i]; ap[8 + i] = a1[i]; }
    }
    __syncthreads();

    // ---- O += P @ G : four 16-wide column chunks, full K=32 each
#pragma unroll
    for (int cc = 0; cc < 4; ++cc) {
      const f16* grow = gB + (cc * 16 + rsel) * 1024 + kt * 32 + (lo ? 0 : 8);
      const v8h g0 = *(const v8h*)(grow);
      const v8h g1 = *(const v8h*)(grow + 16);
      v16h bg;
#pragma unroll
      for (int i = 0; i < 8; ++i) { bg[i] = g0[i]; bg[8 + i] = g1[i]; }
      O[cc] = __builtin_amdgcn_wmma_f32_16x16x32_f16(
          false, ap, false, bg, (short)0, O[cc], false, false);
    }
  }

  // ---- normalize and stage attn_g tile to LDS as f16
#pragma unroll
  for (int r = 0; r < 8; ++r) {
    const float inv = 1.0f / lrow[r];
    const int M = r + (lo ? 0 : 8);
#pragma unroll
    for (int cc = 0; cc < 4; ++cc)
      Obuf[M * 64 + cc * 16 + rsel] = (f16)(O[cc][r] * inv);
  }
  __syncthreads();

  // ---- projection A fragments: attn_g[16x64] as two K=32 chunks
  v16h ao[2];
#pragma unroll
  for (int kc = 0; kc < 2; ++kc) {
    const v8h o0 = *(const v8h*)(&Obuf[rsel * 64 + kc * 32 + (lo ? 0 : 8)]);
    const v8h o1 =
        *(const v8h*)(&Obuf[rsel * 64 + kc * 32 + 16 + (lo ? 0 : 8)]);
#pragma unroll
    for (int i = 0; i < 8; ++i) { ao[kc][i] = o0[i]; ao[kc][8 + i] = o1[i]; }
  }

  const float sigma = sigmaPtr[0];
#pragma unroll
  for (int nc = 0; nc < 8; ++nc) {
    v8f acc = zero8f();
#pragma unroll
    for (int kc = 0; kc < 2; ++kc) {
      const f16* wrow =
          wAT + (nc * 16 + rsel) * 64 + kc * 32 + (lo ? 0 : 8);
      const v8h w0 = *(const v8h*)(wrow);
      const v8h w1 = *(const v8h*)(wrow + 16);
      v16h bw;
#pragma unroll
      for (int i = 0; i < 8; ++i) { bw[i] = w0[i]; bw[8 + i] = w1[i]; }
      acc = __builtin_amdgcn_wmma_f32_16x16x32_f16(
          false, ao[kc], false, bw, (short)0, acc, false, false);
    }
#pragma unroll
    for (int r = 0; r < 8; ++r) {
      const long row = (long)b * 4096 + Mb + r + (lo ? 0 : 8);
      const int col = nc * 16 + rsel;
      out[row * 128 + col] = sigma * acc[r] + x[row * 128 + col];
    }
  }
}

// ---------------------------------------------------------------------------
extern "C" void kernel_launch(void* const* d_in, const int* in_sizes, int n_in,
                              void* d_out, int out_size, void* d_ws,
                              size_t ws_size, hipStream_t stream) {
  (void)in_sizes; (void)n_in; (void)out_size; (void)ws_size;
  const float* x  = (const float*)d_in[0];
  const float* wt = (const float*)d_in[1];
  const float* wp = (const float*)d_in[2];
  const float* wg = (const float*)d_in[3];
  const float* wa = (const float*)d_in[4];
  const float* sg = (const float*)d_in[5];
  float* out = (float*)d_out;

  char* ws = (char*)d_ws;                       // all offsets 256B-aligned
  f16* wTqkg   = (f16*)(ws);                    //    24,576 B
  f16* wAT     = (f16*)(ws + 24576);            //    16,384 B
  f16* projAll = (f16*)(ws + 40960);            // 12,582,912 B
  f16* phiP    = (f16*)(ws + 40960 + 12582912);           //   524,288 B
  f16* gT      = (f16*)(ws + 40960 + 12582912 + 524288);  // 2,097,152 B

  wconv<<<80, 256, 0, stream>>>(wt, wp, wg, wa, wTqkg, wAT);
  proj_qkg<<<4096, 32, 0, stream>>>(x, wTqkg, projAll);
  pool_kg<<<5120, 256, 0, stream>>>(projAll, phiP, gT);
  attn_fuse<<<4096, 32, 0, stream>>>(x, projAll, phiP, gT, wAT, sg, out);
}
